// VanillaRNN_12412455485765
// MI455X (gfx1250) — matrix operand order repaired and checked by reading, not compiled, not run on hardware
//
#include <hip/hip_runtime.h>
#include <math.h>

// ---------------------------------------------------------------------------
// Vanilla RNN on MI455X (gfx1250, wave32, WMMA).
//   h stored transposed as hT[B][H] (bf16) so the per-step GEMM
//   hTnext = tanh( hT @ W_hh^T + x_t (x) W_hx^T + b_h )
//   needs only contiguous row loads for BOTH WMMA operands.
// Precision: bf16 A/B operands, f32 WMMA accumulation (v_wmma_f32_16x16x32_bf16).
// ---------------------------------------------------------------------------

typedef __attribute__((ext_vector_type(16))) __bf16 v16bf;
typedef __attribute__((ext_vector_type(8)))  __bf16 v8bf;
typedef __attribute__((ext_vector_type(4)))  __bf16 v4bf;
typedef __attribute__((ext_vector_type(8)))  float  v8f;

#define RNN_H 2048
#define RNN_B 2048
#define RNN_T 64
#define RNN_C 10

// ---- one-time convert W_hh f32 -> bf16 (stays L2-resident: 8 MB) ----------
__global__ void cvt_w_kernel(const float* __restrict__ w, __bf16* __restrict__ wbf) {
    int i = blockIdx.x * blockDim.x + threadIdx.x;      // one float4 per thread
    float4 f = reinterpret_cast<const float4*>(w)[i];
    v4bf o;
    o[0] = (__bf16)f.x; o[1] = (__bf16)f.y; o[2] = (__bf16)f.z; o[3] = (__bf16)f.w;
    *reinterpret_cast<v4bf*>(wbf + (size_t)i * 4) = o;
}

// ---- h0 = 0 ---------------------------------------------------------------
__global__ void zero_h_kernel(__bf16* __restrict__ h) {
    int i = blockIdx.x * blockDim.x + threadIdx.x;      // 8 bf16 per thread
    v8bf z = {};
    *reinterpret_cast<v8bf*>(h + (size_t)i * 8) = z;
}

// ---------------------------------------------------------------------------
// One recurrence step.
//   D[b, m] = sum_k hT[b, k] * W[m, k]          (GEMM, M'=batch, N'=hidden)
//   hTnext[b, m] = tanh(D + W_hx[m]*x[b,t] + b_h[m])
// Block: 256 threads = 8 waves; block tile 128(batch) x 128(hidden);
// each wave: 32x64 register tile = 2x4 accumulators of 16x16 f32.
// ---------------------------------------------------------------------------
__global__ __launch_bounds__(256)
void rnn_step_kernel(const __bf16* __restrict__ wbf,    // [H][H] W_hh row-major
                     const __bf16* __restrict__ hcur,   // [B][H] hT
                     __bf16*       __restrict__ hnext,  // [B][H]
                     const float*  __restrict__ W_hx,   // [H]
                     const float*  __restrict__ x,      // [B][T]
                     const float*  __restrict__ b_h,    // [H]
                     int t)
{
    const int lane   = threadIdx.x & 31;
    const int wave   = threadIdx.x >> 5;
    const int wm     = wave & 3;                         // 4 waves along batch
    const int wn     = wave >> 2;                        // 2 waves along hidden
    const int mBase  = blockIdx.y * 128 + wm * 32;       // batch rows
    const int nBase  = blockIdx.x * 128 + wn * 64;       // hidden cols
    const int l16    = lane & 15;
    const int hiHalf = lane >> 4;                        // 0 or 1

    v8f acc[2][4];
    #pragma unroll
    for (int i = 0; i < 2; ++i)
        #pragma unroll
        for (int j = 0; j < 4; ++j) acc[i][j] = (v8f){};

    // A rows (hT): lane = batch row within 16-row subtile.
    const __bf16* aRow0 = hcur + (size_t)(mBase +      l16) * RNN_H;
    const __bf16* aRow1 = hcur + (size_t)(mBase + 16 + l16) * RNN_H;
    // B rows (W_hh row m supplies column m of W_hh^T): lane = hidden col.
    const __bf16* bRow0 = wbf + (size_t)(nBase +      l16) * RNN_H;
    const __bf16* bRow1 = wbf + (size_t)(nBase + 16 + l16) * RNN_H;
    const __bf16* bRow2 = wbf + (size_t)(nBase + 32 + l16) * RNN_H;
    const __bf16* bRow3 = wbf + (size_t)(nBase + 48 + l16) * RNN_H;

    const int aOff = hiHalf * 16;   // A 16x32 layout: lanes16-31 hold K=16-31
    const int bOff = hiHalf * 16;   // B 32x16, same k order as A: lanes16-31 hold K=16-31

    union Frag { v16bf v; v8bf h[2]; };

    for (int k0 = 0; k0 < RNN_H; k0 += 32) {
        Frag af[2], bf[4];
        // A fragment: 32 contiguous bytes (K = aOff..aOff+15)
        af[0].h[0] = *reinterpret_cast<const v8bf*>(aRow0 + k0 + aOff);
        af[0].h[1] = *reinterpret_cast<const v8bf*>(aRow0 + k0 + aOff + 8);
        af[1].h[0] = *reinterpret_cast<const v8bf*>(aRow1 + k0 + aOff);
        af[1].h[1] = *reinterpret_cast<const v8bf*>(aRow1 + k0 + aOff + 8);
        // B fragment: 32 contiguous bytes (K = bOff..bOff+15)
        bf[0].h[0] = *reinterpret_cast<const v8bf*>(bRow0 + k0 + bOff);
        bf[0].h[1] = *reinterpret_cast<const v8bf*>(bRow0 + k0 + bOff + 8);
        bf[1].h[0] = *reinterpret_cast<const v8bf*>(bRow1 + k0 + bOff);
        bf[1].h[1] = *reinterpret_cast<const v8bf*>(bRow1 + k0 + bOff + 8);
        bf[2].h[0] = *reinterpret_cast<const v8bf*>(bRow2 + k0 + bOff);
        bf[2].h[1] = *reinterpret_cast<const v8bf*>(bRow2 + k0 + bOff + 8);
        bf[3].h[0] = *reinterpret_cast<const v8bf*>(bRow3 + k0 + bOff);
        bf[3].h[1] = *reinterpret_cast<const v8bf*>(bRow3 + k0 + bOff + 8);

        #pragma unroll
        for (int i = 0; i < 2; ++i)
            #pragma unroll
            for (int j = 0; j < 4; ++j)
                acc[i][j] = __builtin_amdgcn_wmma_f32_16x16x32_bf16(
                    /*neg_a=*/false, af[i].v, /*neg_b=*/false, bf[j].v,
                    /*c_mod=*/(short)0, acc[i][j],
                    /*reuse_a=*/false, /*reuse_b=*/false);
    }

    // Epilogue: rank-1 input term + bias + tanh, store bf16.
    // D tile layout: VGPR r -> batch row (r + 8*hiHalf), N col = l16.
    float xv[2][8];
    #pragma unroll
    for (int i = 0; i < 2; ++i)
        #pragma unroll
        for (int r = 0; r < 8; ++r)
            xv[i][r] = x[(size_t)(mBase + i * 16 + hiHalf * 8 + r) * RNN_T + t];

    #pragma unroll
    for (int j = 0; j < 4; ++j) {
        const int hid = nBase + j * 16 + l16;
        const float whx = W_hx[hid];
        const float bh  = b_h[hid];
        #pragma unroll
        for (int i = 0; i < 2; ++i) {
            const int bRowIdx = mBase + i * 16 + hiHalf * 8;
            #pragma unroll
            for (int r = 0; r < 8; ++r) {
                float v = acc[i][j][r] + whx * xv[i][r] + bh;
                hnext[(size_t)(bRowIdx + r) * RNN_H + hid] = (__bf16)tanhf(v);
            }
        }
    }
}

// ---- out[b, c] = W_hy[:,c] . hT[b,:] + b_p[c]; one wave per batch row -----
__global__ __launch_bounds__(32)
void out_proj_kernel(const __bf16* __restrict__ hT,     // [B][H]
                     const float*  __restrict__ W_hy,   // [H][C]
                     const float*  __restrict__ b_p,    // [C]
                     float*        __restrict__ out)    // [B][C]
{
    const int b    = blockIdx.x;
    const int lane = threadIdx.x;
    float acc[RNN_C];
    #pragma unroll
    for (int c = 0; c < RNN_C; ++c) acc[c] = 0.0f;

    const __bf16* hrow = hT + (size_t)b * RNN_H;
    for (int h = lane; h < RNN_H; h += 32) {
        const float hv = (float)hrow[h];
        const float* wr = W_hy + (size_t)h * RNN_C;
        #pragma unroll
        for (int c = 0; c < RNN_C; ++c) acc[c] += hv * wr[c];
    }
    #pragma unroll
    for (int c = 0; c < RNN_C; ++c) {
        float v = acc[c];
        for (int off = 16; off > 0; off >>= 1) v += __shfl_xor(v, off, 32);
        if (lane == 0) out[(size_t)b * RNN_C + c] = v + b_p[c];
    }
}

// ---------------------------------------------------------------------------
extern "C" void kernel_launch(void* const* d_in, const int* in_sizes, int n_in,
                              void* d_out, int out_size, void* d_ws, size_t ws_size,
                              hipStream_t stream) {
    (void)in_sizes; (void)n_in; (void)out_size; (void)ws_size;
    const float* x    = (const float*)d_in[0];   // [B,T]
    const float* W_hx = (const float*)d_in[1];   // [H,1]
    const float* W_hh = (const float*)d_in[2];   // [H,H]
    const float* W_hy = (const float*)d_in[3];   // [H,C]
    const float* b_h  = (const float*)d_in[4];   // [H,1]
    const float* b_p  = (const float*)d_in[5];   // [C,1]
    float* out = (float*)d_out;

    // Workspace: W_hh(bf16) 8MB | hT ping 8MB | hT pong 8MB  (24 MB total)
    char* ws = (char*)d_ws;
    __bf16* wbf = (__bf16*)ws;
    __bf16* h0  = (__bf16*)(ws + (size_t)RNN_H * RNN_H * 2);
    __bf16* h1  = (__bf16*)(ws + (size_t)RNN_H * RNN_H * 2 + (size_t)RNN_B * RNN_H * 2);

    cvt_w_kernel<<<(RNN_H * RNN_H / 4) / 256, 256, 0, stream>>>(W_hh, wbf);
    zero_h_kernel<<<(RNN_B * RNN_H / 8) / 256, 256, 0, stream>>>(h0);

    __bf16* cur = h0;
    __bf16* nxt = h1;
    dim3 grid(RNN_H / 128, RNN_B / 128);   // (hidden tiles, batch tiles)
    for (int t = 0; t < RNN_T; ++t) {
        rnn_step_kernel<<<grid, 256, 0, stream>>>(wbf, cur, nxt, W_hx, x, b_h, t);
        __bf16* tmp = cur; cur = nxt; nxt = tmp;
    }

    out_proj_kernel<<<RNN_B, 32, 0, stream>>>(cur, W_hy, b_p, out);
}
